// EGNN_dynamics_49976239456426
// MI455X (gfx1250) — matrix operand order repaired and checked by reading, not compiled, not run on hardware
//
#include <hip/hip_runtime.h>
#include <hip/hip_fp16.h>

// ---------------------------------------------------------------------------
// EGNN forward on gfx1250 (CDNA5).
// Block-tiled WMMA GEMMs: 8-wave block -> 128x256 C tile, 16 accumulators per
// wave (A-fragment reuse x16), B double-buffered in LDS via
// global_load_async_to_lds_b128 (ASYNCcnt), fp16 data / fp32 accumulate.
// ---------------------------------------------------------------------------

typedef __attribute__((ext_vector_type(16))) _Float16 v16h;
typedef __attribute__((ext_vector_type(8)))  float    v8f;

#define NN_NODES   1536      // 32*48
#define NE_EDGES   73728     // 32*48*48
#define NPG        48        // nodes per graph (dense edge block)
#define HD         256       // hidden
#define KEDGE      544       // 2*256+2 padded up to 17*32
#define KNODE      512
#define BROW       40        // LDS B row stride in halves (80B: bank-conflict free)
#define COORDS_RANGE 3.75f   // 15/4
#define INV_NORM   0.01f     // 1/100

__device__ __forceinline__ float silu_f(float x) { return x / (1.f + __expf(-x)); }

__device__ __forceinline__ v16h load_a_frag(const _Float16* __restrict__ ap)
{
    v16h a;
#pragma unroll
    for (int v = 0; v < 4; ++v) {             // lo half-wave K {0..7,16..23}; hi +8
        a[2*v]   = ap[2*v];      a[2*v+1] = ap[2*v+1];
        a[8+2*v] = ap[16+2*v];   a[9+2*v] = ap[17+2*v];
    }
    return a;
}

// ---------------- weight convert: fp32 [K,N] -> fp16 transposed [Npad,Kpad] --
__global__ void cvt_w_kernel(const float* __restrict__ W, _Float16* __restrict__ Bt,
                             int K, int N, int Kpad, int total)
{
    int idx = blockIdx.x * blockDim.x + threadIdx.x;
    if (idx >= total) return;
    int n = idx / Kpad, k = idx % Kpad;
    float v = (k < K && n < N) ? W[(size_t)k * N + n] : 0.f;
    Bt[idx] = (_Float16)v;
}

// ---------------------------------------------------------------------------
// Big block-tiled GEMM, N fixed = 256. Block = 8 waves, C tile = 128x256.
// EDGE=true: A row e = [h[row[e]](256) | h[col[e]](256) | d2 | dtop | 0pad].
// B chunk [256 cols x 32 K] staged to LDS with async-to-LDS, double-buffered.
// ---------------------------------------------------------------------------
template<bool EDGE>
__global__ void gemm_big(const _Float16* __restrict__ A, int Kpad,
                         const _Float16* __restrict__ hbuf,
                         const int* __restrict__ row, const int* __restrict__ col,
                         const _Float16* __restrict__ d2cur,
                         const _Float16* __restrict__ dtop,
                         const _Float16* __restrict__ Bt,
                         const float* __restrict__ bias,
                         _Float16* __restrict__ Ch, float* __restrict__ Cf,
                         const float* __restrict__ rmask, int act)
{
    __shared__ _Float16 Bl[2][256 * BROW];    // 2 x 20KB double buffer

    const int tid  = threadIdx.x;
    const int lane = tid & 31;
    const int wv   = tid >> 5;
    const int mt   = blockIdx.x * 8 + wv;     // this wave's 16-row M tile
    const int sel  = lane >> 4, lm = lane & 15;
    const int rowA = (mt << 4) + lm;

    // async stage of B chunk ks (32 K values, one row per thread) into Bl[ks&1]
    auto stage = [&](int ks) {
        const _Float16* gsrc = Bt + (size_t)tid * Kpad + ks * 32;
        unsigned ldst = (unsigned)(size_t)(&Bl[ks & 1][tid * BROW]);
        asm volatile("global_load_async_to_lds_b128 %0, %1, off\n\t"
                     "global_load_async_to_lds_b128 %0, %1, off offset:16"
                     :: "v"(ldst), "v"(gsrc) : "memory");
    };

    // edge-gather state (dead code when !EDGE)
    const _Float16* hr = nullptr;
    const _Float16* hc = nullptr;
    const _Float16* arow = nullptr;
    _Float16 d2e = (_Float16)0.f, dte = (_Float16)0.f;
    if constexpr (EDGE) {
        const int re = row[rowA], ce = col[rowA];
        hr = hbuf + (size_t)re * HD;
        hc = hbuf + (size_t)ce * HD;
        d2e = d2cur[rowA]; dte = dtop[rowA];
    } else {
        arow = A + (size_t)rowA * Kpad;
    }

    v8f acc[16];
#pragma unroll
    for (int nt = 0; nt < 16; ++nt) acc[nt] = v8f{};

    stage(0);
    asm volatile("s_wait_asynccnt 0" ::: "memory");
    __syncthreads();

    const int nsteps = Kpad / 32;
    for (int ks = 0; ks < nsteps; ++ks) {
        if (ks + 1 < nsteps) stage(ks + 1);   // prefetch next chunk into other buf

        const int k0 = ks * 32;
        v16h a;
        if constexpr (EDGE) {
            if (k0 < KNODE) {
                const _Float16* ap =
                    ((k0 < HD) ? (hr + k0) : (hc + (k0 - HD))) + (sel ? 8 : 0);
                a = load_a_frag(ap);
            } else {                          // k0 == 512: [d2, dist_top, zeros]
#pragma unroll
                for (int i = 0; i < 16; ++i) a[i] = (_Float16)0.f;
                if (!sel) { a[0] = d2e; a[1] = dte; }
            }
        } else {
            a = load_a_frag(arow + k0 + (sel ? 8 : 0));
        }

        const _Float16* bl = &Bl[ks & 1][0];
#pragma unroll
        for (int nt = 0; nt < 16; ++nt) {
            const _Float16* bp = bl + (nt * 16 + lm) * BROW + sel * 8;
            v16h b;
#pragma unroll
            for (int i = 0; i < 16; ++i) b[i] = bp[i + (i >> 3) * 8];
            acc[nt] = __builtin_amdgcn_wmma_f32_16x16x32_f16(false, a, false, b,
                                                             (short)0, acc[nt],
                                                             false, false);
        }

        asm volatile("s_wait_asynccnt 0" ::: "memory");
        __syncthreads();                      // staged chunk visible; safe to reuse buf
    }

#pragma unroll
    for (int nt = 0; nt < 16; ++nt) {
#pragma unroll
        for (int v = 0; v < 8; ++v) {
            const int r = (mt << 4) + v + (sel ? 8 : 0);
            const int c = (nt << 4) + lm;
            float val = acc[nt][v] + (bias ? bias[c] : 0.f);
            if (act)   val = silu_f(val);
            if (rmask) val *= rmask[r];
            if (Ch) Ch[(size_t)r * HD + c] = (_Float16)val;
            if (Cf) Cf[(size_t)r * HD + c] = val;
        }
    }
}

// ---------------- small per-wave GEMM (only for the N=6 output layer) -------
__global__ void gemm_wmma_small(const _Float16* __restrict__ A,
                                const _Float16* __restrict__ Bt,
                                const float* __restrict__ bias,
                                float* __restrict__ Cf,
                                int M, int Kpad, int Nreal, int ldc)
{
    const int lane = threadIdx.x & 31;
    const int wave = blockIdx.x * (blockDim.x >> 5) + (threadIdx.x >> 5);
    if (wave >= (M >> 4)) return;
    const int mt = wave;
    const int sel = lane >> 4, lm = lane & 15;

    const _Float16* arow = A  + (size_t)((mt << 4) + lm) * Kpad;
    const _Float16* brow = Bt + (size_t)lm * Kpad;

    v8f acc = {};
    for (int k0 = 0; k0 < Kpad; k0 += 32) {
        v16h a = load_a_frag(arow + k0 + (sel ? 8 : 0));
        const _Float16* bp = brow + k0 + (sel ? 8 : 0);
        v16h b;
#pragma unroll
        for (int i = 0; i < 16; ++i) b[i] = bp[i + (i >> 3) * 8];
        acc = __builtin_amdgcn_wmma_f32_16x16x32_f16(false, a, false, b,
                                                     (short)0, acc, false, false);
    }
#pragma unroll
    for (int v = 0; v < 8; ++v) {
        const int r = (mt << 4) + v + (sel ? 8 : 0);
        const int c = lm;
        if (c < Nreal) Cf[(size_t)r * ldc + c] = acc[v] + bias[c];
    }
}

// ---------------- node prep: x0, x, padded fp16 h-input [nn,32] -------------
__global__ void prep_kernel(const float* __restrict__ t, const float* __restrict__ xh,
                            const float* __restrict__ nm,
                            float* __restrict__ x, float* __restrict__ x0,
                            _Float16* __restrict__ hin)
{
    int n = blockIdx.x * blockDim.x + threadIdx.x;
    if (n >= NN_NODES) return;
    float m = nm[n];
    int b = n / NPG;
#pragma unroll
    for (int d = 0; d < 3; ++d) { float v = xh[n*8 + d] * m; x[n*3 + d] = v; x0[n*3 + d] = v; }
#pragma unroll
    for (int k = 0; k < 32; ++k) {
        float v = 0.f;
        if (k < 5) v = xh[n*8 + 3 + k] * m;
        else if (k == 5) v = t[b];            // time feature, unmasked per reference
        hin[n*32 + k] = (_Float16)v;
    }
}

// ---------------- edge geometry: d2, coord_diff = diff / (sqrt(d2+eps)+1) ----
__global__ void edge_geom_kernel(const float* __restrict__ x,
                                 const int* __restrict__ row, const int* __restrict__ col,
                                 _Float16* __restrict__ d2out, float* __restrict__ cdiff)
{
    int e = blockIdx.x * blockDim.x + threadIdx.x;
    if (e >= NE_EDGES) return;
    int r = row[e], c = col[e];
    float dx = x[r*3+0] - x[c*3+0];
    float dy = x[r*3+1] - x[c*3+1];
    float dz = x[r*3+2] - x[c*3+2];
    float d2 = dx*dx + dy*dy + dz*dz;
    float inv = 1.f / (sqrtf(d2 + 1e-8f) + 1.f);
    cdiff[e*3+0] = dx * inv; cdiff[e*3+1] = dy * inv; cdiff[e*3+2] = dz * inv;
    d2out[e] = (_Float16)d2;
}

// ---------------- dense segment-sum (48 contiguous edges per node) + concat --
__global__ void node_agg_concat(const _Float16* __restrict__ m,
                                const _Float16* __restrict__ h,
                                _Float16* __restrict__ node_in)
{
    int n = blockIdx.x, c = threadIdx.x;      // 256 threads = channels
    const _Float16* p = m + (size_t)n * NPG * HD + c;
    float s = 0.f;
    for (int j = 0; j < NPG; ++j) s += (float)p[(size_t)j * HD];
    node_in[(size_t)n * KNODE + c]      = h[(size_t)n * HD + c];
    node_in[(size_t)n * KNODE + HD + c] = (_Float16)(s * INV_NORM);
}

// ---------------- h = (h + upd) * nm ----------------------------------------
__global__ void h_update_kernel(_Float16* __restrict__ h, const float* __restrict__ upd,
                                const float* __restrict__ nm)
{
    int n = blockIdx.x, c = threadIdx.x;
    size_t i = (size_t)n * HD + c;
    h[i] = (_Float16)(((float)h[i] + upd[i]) * nm[n]);
}

__global__ void mask_h_kernel(_Float16* __restrict__ h, const float* __restrict__ nm)
{
    int n = blockIdx.x, c = threadIdx.x;
    size_t i = (size_t)n * HD + c;
    h[i] = (_Float16)((float)h[i] * nm[n]);
}

// ---------------- coord: phi = c1out . c2; x += segsum(cdiff*tanh(phi)*...) --
__global__ void coord_apply_kernel(const _Float16* __restrict__ c1out,
                                   const float* __restrict__ c2,
                                   const float* __restrict__ cdiff,
                                   const float* __restrict__ emask,
                                   const int* __restrict__ row,
                                   float* __restrict__ x)
{
    int wave = blockIdx.x * (blockDim.x >> 5) + (threadIdx.x >> 5);
    int lane = threadIdx.x & 31;
    if (wave >= NE_EDGES) return;
    const _Float16* p = c1out + (size_t)wave * HD;
    float s = 0.f;
#pragma unroll
    for (int i = 0; i < HD / 32; ++i) s += (float)p[lane + 32*i] * c2[lane + 32*i];
    for (int off = 16; off > 0; off >>= 1) s += __shfl_down(s, off, 32);
    float phi = __shfl(s, 0, 32);
    if (lane < 3) {
        float g = tanhf(phi) * COORDS_RANGE * emask[wave] * INV_NORM;
        atomicAdd(&x[(size_t)row[wave] * 3 + lane], cdiff[(size_t)wave * 3 + lane] * g);
    }
}

// ---------------- final: [vel, h[:, :5]] ------------------------------------
__global__ void final_kernel(const float* __restrict__ x, const float* __restrict__ x0,
                             const float* __restrict__ hf, const float* __restrict__ nm,
                             float* __restrict__ out)
{
    int n = blockIdx.x * blockDim.x + threadIdx.x;
    if (n >= NN_NODES) return;
    float m = nm[n];
#pragma unroll
    for (int d = 0; d < 3; ++d) out[n*8 + d] = (x[n*3 + d] - x0[n*3 + d]) * m;
#pragma unroll
    for (int k = 0; k < 5; ++k) out[n*8 + 3 + k] = hf[n*6 + k] * m;
}

// ============================================================================
extern "C" void kernel_launch(void* const* d_in, const int* in_sizes, int n_in,
                              void* d_out, int out_size, void* d_ws, size_t ws_size,
                              hipStream_t stream)
{
    (void)in_sizes; (void)n_in; (void)out_size; (void)ws_size;
    // setup_inputs() dict order: t, xh, node_mask, edge_mask, edge_index, params(leaves)
    const float* t_in  = (const float*)d_in[0];
    const float* xh    = (const float*)d_in[1];
    const float* nmask = (const float*)d_in[2];
    const float* emask = (const float*)d_in[3];
    const int*   eidx  = (const int*)d_in[4];
    const int* row = eidx;
    const int* col = eidx + NE_EDGES;
    const int PB = 5;  // params leaves: emb(2 lin), emb_out(3 lin), blocks[4]{gcls[2]{e0,e1,n0,n1}, coord{c0,c1,c2}}

    char* wp = (char*)d_ws;
    auto alloc = [&](size_t bytes) -> void* {
        void* r = (void*)wp; wp += (bytes + 255) & ~(size_t)255; return r;
    };
    auto cvtW = [&](int idx, int K, int N, int Kpad, int Npad) -> _Float16* {
        _Float16* bt = (_Float16*)alloc((size_t)Kpad * Npad * sizeof(_Float16));
        int tot = Kpad * Npad;
        cvt_w_kernel<<<(tot + 255) / 256, 256, 0, stream>>>(
            (const float*)d_in[idx], bt, K, N, Kpad, tot);
        return bt;
    };

    // ---- convert all weights once (transposed fp16) ----
    _Float16* W_emb0 = cvtW(PB + 0, 6, 256, 32, 256);
    _Float16* W_emb1 = cvtW(PB + 2, 256, 256, 256, 256);
    _Float16* W_o0   = cvtW(PB + 4, 256, 256, 256, 256);
    _Float16* W_o1   = cvtW(PB + 6, 256, 256, 256, 256);
    _Float16* W_o2   = cvtW(PB + 8, 256, 6, 256, 16);
    const float *b_emb0 = (const float*)d_in[PB+1], *b_emb1 = (const float*)d_in[PB+3];
    const float *b_o0 = (const float*)d_in[PB+5], *b_o1 = (const float*)d_in[PB+7],
                *b_o2 = (const float*)d_in[PB+9];

    _Float16 *We0[4][2], *We1[4][2], *Wn0[4][2], *Wn1[4][2], *Wc0[4], *Wc1[4];
    const float *be0[4][2], *be1[4][2], *bn0[4][2], *bn1[4][2], *bc0[4], *bc1[4], *c2p[4];
    for (int b = 0; b < 4; ++b) {
        int base = PB + 10 + b * 21;
        for (int g = 0; g < 2; ++g) {
            int gb = base + g * 8;
            We0[b][g] = cvtW(gb + 0, 514, 256, KEDGE, 256); be0[b][g] = (const float*)d_in[gb + 1];
            We1[b][g] = cvtW(gb + 2, 256, 256, 256, 256);   be1[b][g] = (const float*)d_in[gb + 3];
            Wn0[b][g] = cvtW(gb + 4, 512, 256, 512, 256);   bn0[b][g] = (const float*)d_in[gb + 5];
            Wn1[b][g] = cvtW(gb + 6, 256, 256, 256, 256);   bn1[b][g] = (const float*)d_in[gb + 7];
        }
        int cb = base + 16;
        Wc0[b] = cvtW(cb + 0, 514, 256, KEDGE, 256); bc0[b] = (const float*)d_in[cb + 1];
        Wc1[b] = cvtW(cb + 2, 256, 256, 256, 256);   bc1[b] = (const float*)d_in[cb + 3];
        c2p[b] = (const float*)d_in[cb + 4];
    }

    // ---- activation buffers ----
    _Float16* h16     = (_Float16*)alloc((size_t)NN_NODES * HD * 2);
    _Float16* hin16   = (_Float16*)alloc((size_t)NN_NODES * 32 * 2);
    _Float16* nhid    = (_Float16*)alloc((size_t)NN_NODES * HD * 2);
    _Float16* nhid2   = (_Float16*)alloc((size_t)NN_NODES * HD * 2);
    _Float16* node_in = (_Float16*)alloc((size_t)NN_NODES * KNODE * 2);
    float*    nupd    = (float*)   alloc((size_t)NN_NODES * HD * 4);
    float*    x       = (float*)   alloc((size_t)NN_NODES * 3 * 4);
    float*    x0      = (float*)   alloc((size_t)NN_NODES * 3 * 4);
    float*    cdiff   = (float*)   alloc((size_t)NE_EDGES * 3 * 4);
    _Float16* d2cur   = (_Float16*)alloc((size_t)NE_EDGES * 2);
    _Float16* dtop    = (_Float16*)alloc((size_t)NE_EDGES * 2);
    _Float16* ebuf0   = (_Float16*)alloc((size_t)NE_EDGES * HD * 2);
    _Float16* ebuf1   = (_Float16*)alloc((size_t)NE_EDGES * HD * 2);
    float*    houtf   = (float*)   alloc((size_t)NN_NODES * 6 * 4);

    auto gemmBig = [&](const _Float16* A, int Kpad, const _Float16* Bt, const float* bias,
                       _Float16* Ch, float* Cf, const float* rmask, int M, int act) {
        gemm_big<false><<<M / 128, 256, 0, stream>>>(
            A, Kpad, nullptr, nullptr, nullptr, nullptr, nullptr,
            Bt, bias, Ch, Cf, rmask, act);
    };
    auto gemmEdge = [&](const _Float16* Bt, const float* bias, _Float16* Ch,
                        const float* rmask, int act) {
        gemm_big<true><<<NE_EDGES / 128, 256, 0, stream>>>(
            nullptr, KEDGE, h16, row, col, d2cur, dtop,
            Bt, bias, Ch, nullptr, rmask, act);
    };

    // ---- forward ----
    prep_kernel<<<(NN_NODES + 255) / 256, 256, 0, stream>>>(t_in, xh, nmask, x, x0, hin16);
    edge_geom_kernel<<<(NE_EDGES + 255) / 256, 256, 0, stream>>>(x0, row, col, dtop, cdiff);

    gemmBig(hin16, 32,  W_emb0, b_emb0, nhid, nullptr, nullptr, NN_NODES, 1);
    gemmBig(nhid,  256, W_emb1, b_emb1, h16,  nullptr, nullptr, NN_NODES, 0);

    for (int b = 0; b < 4; ++b) {
        edge_geom_kernel<<<(NE_EDGES + 255) / 256, 256, 0, stream>>>(x, row, col, d2cur, cdiff);
        for (int g = 0; g < 2; ++g) {
            gemmEdge(We0[b][g], be0[b][g], ebuf0, nullptr, 1);                  // silu(e0)
            gemmBig(ebuf0, 256, We1[b][g], be1[b][g], ebuf1, nullptr, emask,
                    NE_EDGES, 1);                                               // silu(e1)*emask
            node_agg_concat<<<NN_NODES, 256, 0, stream>>>(ebuf1, h16, node_in);
            gemmBig(node_in, 512, Wn0[b][g], bn0[b][g], nhid, nullptr, nullptr,
                    NN_NODES, 1);                                               // silu(n0)
            gemmBig(nhid, 256, Wn1[b][g], bn1[b][g], nullptr, nupd, nullptr,
                    NN_NODES, 0);                                               // n1 -> fp32 upd
            h_update_kernel<<<NN_NODES, 256, 0, stream>>>(h16, nupd, nmask);
        }
        gemmEdge(Wc0[b], bc0[b], ebuf0, nullptr, 1);                            // silu(c0)
        gemmBig(ebuf0, 256, Wc1[b], bc1[b], ebuf1, nullptr, nullptr,
                NE_EDGES, 1);                                                   // silu(c1)
        coord_apply_kernel<<<NE_EDGES / 8, 256, 0, stream>>>(ebuf1, c2p[b], cdiff,
                                                             emask, row, x);
        mask_h_kernel<<<NN_NODES, 256, 0, stream>>>(h16, nmask);
    }

    gemmBig(h16,  256, W_o0, b_o0, nhid,  nullptr, nullptr, NN_NODES, 1);
    gemmBig(nhid, 256, W_o1, b_o1, nhid2, nullptr, nullptr, NN_NODES, 1);
    gemm_wmma_small<<<(NN_NODES / 16 + 7) / 8, 256, 0, stream>>>(
        nhid2, W_o2, b_o2, houtf, NN_NODES, 256, 6, 6);
    final_kernel<<<(NN_NODES + 255) / 256, 256, 0, stream>>>(x, x0, houtf, nmask,
                                                             (float*)d_out);
}
